// ClusteringVectorQuantiser_43267500540448
// MI455X (gfx1250) — matrix operand order repaired and checked by reading, not compiled, not run on hardware
//
#include <hip/hip_runtime.h>
#include <cstdint>

// ---- problem constants (fixed by the reference) ----
#define D        512
#define KCODES   1024
#define NROWS    65536           // 16*4096
#define BETA     0.25f
#define ROWS_PB  128             // rows per block (8 waves * 16)
#define ESTR_U   264             // uints per LDS code row: (512 bf16 + 16 pad)/2
#define TILE_U   4224            // 16 * ESTR_U

typedef __attribute__((ext_vector_type(16))) __bf16   v16bf;
typedef __attribute__((ext_vector_type(8)))  float    v8f;
typedef __attribute__((ext_vector_type(4)))  unsigned u32x4;
typedef __attribute__((ext_vector_type(8)))  int      i32x8;
typedef __attribute__((ext_vector_type(4)))  int      i32x4;

__device__ __forceinline__ unsigned f2bf(float f) {
    union { float f; unsigned u; } v; v.f = f;
    unsigned u = v.u;
    return (u + 0x7FFFu + ((u >> 16) & 1u)) >> 16;   // RNE truncate to bf16
}

union Frag { u32x4 q[2]; v16bf v; };

#if __has_builtin(__builtin_amdgcn_tensor_load_to_lds) && \
    __has_builtin(__builtin_amdgcn_s_wait_tensorcnt)
#define USE_TDM 1
#else
#define USE_TDM 0
#endif

#if USE_TDM
// TDM: DMA one 16x512 bf16 code tile (row stride 512 elems) into LDS with
// 8-dword padding every 256 dwords -> padded LDS stride of 264 uints.
__device__ __forceinline__ void tdm_load_tile(const unsigned* gsrc, unsigned lds_off) {
    unsigned long ga = (unsigned long)gsrc;
    u32x4 g0;
    g0[0] = 1u;                                                   // count=1 (valid user D#)
    g0[1] = lds_off;                                              // lds_addr (bytes)
    g0[2] = (unsigned)ga;                                         // global_addr[31:0]
    g0[3] = (unsigned)((ga >> 32) & 0x01FFFFFFu) | (2u << 30);    // global_addr[56:32], type=2
    i32x8 g1;
    g1[0] = (int)((1u << 16)      // data_size = 2 bytes
                | (1u << 20)      // pad_enable
                | (7u << 22)      // pad_interval = 256 dwords (one 512-bf16 row)
                | (7u << 25));    // pad_amount   = 8 dwords (16 bf16)
    g1[1] = (int)(512u << 16);    // atomic_barrier_addr=0 | tensor_dim0[15:0]=512
    g1[2] = (int)(1024u << 16);   // tensor_dim0[31:16]=0  | tensor_dim1[15:0]=1024
    g1[3] = (int)(512u << 16);    // tensor_dim1[31:16]=0  | tile_dim0=512
    g1[4] = 16;                   // tile_dim1=16 | tile_dim2=0
    g1[5] = 512;                  // tensor_dim0_stride[31:0]=512 (elements)
    g1[6] = 0;                    // stride0[47:32]=0 | tensor_dim1_stride[15:0]=0
    g1[7] = 0;
    i32x4 z4 = {0, 0, 0, 0};
#if __has_include(<hip/amd_detail/amd_gfx1250_TDM.h>)
    i32x8 z8 = {0, 0, 0, 0, 0, 0, 0, 0};
    __builtin_amdgcn_tensor_load_to_lds(g0, g1, z4, z4, z8, 0);   // therock 6-arg form
#else
    __builtin_amdgcn_tensor_load_to_lds(g0, g1, z4, z4, 0);       // ROCm 7.2 5-arg form
#endif
}
#endif

// ---------------- kernel 1: codebook fp32 -> bf16 + |e_k|^2 ----------------
__global__ void vq_prep(const float* __restrict__ e,
                        unsigned* __restrict__ ebf,     // [1024][256] uints (2 bf16 each)
                        float* __restrict__ ssq) {      // [1024]
    const int k = blockIdx.x;        // code id
    const int t = threadIdx.x;       // 0..127
    float4 v = ((const float4*)(e + k * D))[t];
    float s = v.x*v.x + v.y*v.y + v.z*v.z + v.w*v.w;
    ebf[k*256 + t*2]     = f2bf(v.x) | (f2bf(v.y) << 16);
    ebf[k*256 + t*2 + 1] = f2bf(v.z) | (f2bf(v.w) << 16);
    __shared__ float red[128];
    red[t] = s; __syncthreads();
    for (int o = 64; o > 0; o >>= 1) { if (t < o) red[t] += red[t+o]; __syncthreads(); }
    if (t == 0) ssq[k] = red[0];
}

// ---------------- kernel 2: WMMA score GEMM + argmax ----------------
// B (z rows, bf16) held entirely in registers; code tiles double-buffered in
// LDS via the Tensor Data Mover; scores s = 2*z.e - |e|^2; per-lane argmax.
__global__ void __launch_bounds__(256)
vq_argmax(const float* __restrict__ z,
          const unsigned* __restrict__ ebf,
          const float* __restrict__ ssq,
          int* __restrict__ idx_out) {
    __shared__ unsigned et[2 * TILE_U];   // two 16x264-uint code-tile buffers
    __shared__ float    ssqs[KCODES];

    const int t    = threadIdx.x;
    const int lane = t & 31;
    const int wave = t >> 5;
    const int mrow = lane & 15;   // z-row within wave tile == code row in tile == C column
    const int half = lane >> 4;
    const long rowbase = (long)blockIdx.x * ROWS_PB;

    // stage |e|^2 into LDS (published by first in-loop barrier)
    #pragma unroll
    for (int i = 0; i < 4; ++i) ssqs[t + i * 256] = ssq[t + i * 256];

    // ---- preload this wave's B fragments (16 z rows x 512 k) into registers ----
    // B 32x16 bf16 layout: lane(half=0) holds K=0..15, lane(half=1) K=16..31.
    Frag bfr[16];
    {
        const float* zrow = z + (rowbase + wave * 16 + mrow) * (long)D + half * 16;
        #pragma unroll
        for (int ks = 0; ks < 16; ++ks) {
            const float4* p = (const float4*)(zrow + ks * 32);
            float4 f0 = p[0], f1 = p[1], f2 = p[2], f3 = p[3];
            u32x4 lo = { f2bf(f0.x) | (f2bf(f0.y) << 16), f2bf(f0.z) | (f2bf(f0.w) << 16),
                         f2bf(f1.x) | (f2bf(f1.y) << 16), f2bf(f1.z) | (f2bf(f1.w) << 16) };
            u32x4 hi = { f2bf(f2.x) | (f2bf(f2.y) << 16), f2bf(f2.z) | (f2bf(f2.w) << 16),
                         f2bf(f3.x) | (f2bf(f3.y) << 16), f2bf(f3.z) | (f2bf(f3.w) << 16) };
            bfr[ks].q[0] = lo;
            bfr[ks].q[1] = hi;
        }
    }

#if USE_TDM
    if (t < 32) tdm_load_tile(ebf, (unsigned)(size_t)&et[0]);     // prefetch tile 0
#endif

    float best = -3.4e38f;
    int   bidx = 0;

    for (int ct = 0; ct < 64; ++ct) {
        const unsigned* cur = et + (ct & 1) * TILE_U;
#if USE_TDM
        if (t < 32) {
            if (ct < 63) {   // prefetch next tile into the other buffer
                tdm_load_tile(ebf + (ct + 1) * 4096,
                              (unsigned)(size_t)&et[((ct + 1) & 1) * TILE_U]);
                __builtin_amdgcn_s_wait_tensorcnt(1);   // current tile complete
            } else {
                __builtin_amdgcn_s_wait_tensorcnt(0);
            }
        }
        __syncthreads();   // publish current tile to all waves
#else
        __syncthreads();   // previous use of this buffer finished
        #pragma unroll
        for (int i = 0; i < 16; ++i) {
            int l = t + i * 256;          // 0..4095
            int r = l >> 8, c = l & 255;
            ((unsigned*)cur)[r * ESTR_U + c] = ebf[(ct * 16 + r) * 256 + c];
        }
        __syncthreads();
#endif
        const unsigned* arow = cur + mrow * ESTR_U;

        v8f acc = {0.f,0.f,0.f,0.f,0.f,0.f,0.f,0.f};
        #pragma unroll
        for (int ks = 0; ks < 16; ++ks) {
            // A 16x32 bf16: K = half*16 + {0..15}
            Frag a;
            a.q[0] = *(const u32x4*)(arow + ks * 16 + half * 8);
            a.q[1] = *(const u32x4*)(arow + ks * 16 + half * 8 + 4);
            acc = __builtin_amdgcn_wmma_f32_16x16x32_bf16(
                      false, a.v, false, bfr[ks].v, (short)0, acc, false, false);
        }

        // score = 2*(z.e) - |e|^2 ; lane holds codes (half*8 + vv) for z-row (lane&15)
        const int cbase = ct * 16 + half * 8;
        #pragma unroll
        for (int vv = 0; vv < 8; ++vv) {
            int code = cbase + vv;
            float s = 2.0f * acc[vv] - ssqs[code];
            if (s > best) { best = s; bidx = code; }   // strict > keeps first max
        }
        __syncthreads();   // all waves done reading 'cur' before it is refilled
    }

    // combine the two half-lane candidate sets for this z row (lane n <-> n+16)
    float ob = __shfl_xor(best, 16, 32);
    int   oi = __shfl_xor(bidx, 16, 32);
    if (ob > best || (ob == best && oi < bidx)) { best = ob; bidx = oi; }
    if (half == 0) idx_out[rowbase + wave * 16 + mrow] = bidx;
}

// ---------------- kernel 3: gather z_q, write outputs, per-row loss sums ----------------
__global__ void vq_gather(const float* __restrict__ z,
                          const float* __restrict__ e,
                          const int* __restrict__ idx,
                          float* __restrict__ zq_out,   // d_out[0 .. N*D)
                          float* __restrict__ idx_out,  // d_out[N*D+1 ..)
                          float* __restrict__ rowsum) { // ws
    const int row = blockIdx.x;
    const int t   = threadIdx.x;      // 0..127
    const int code = idx[row];
    float4 ev = ((const float4*)(e + (long)code * D))[t];
    float4 zv = ((const float4*)(z + (long)row  * D))[t];
    ((float4*)(zq_out + (long)row * D))[t] = ev;   // z_q_st == z_q value-wise
    float dx = ev.x - zv.x, dy = ev.y - zv.y, dz = ev.z - zv.z, dw = ev.w - zv.w;
    float s = dx*dx + dy*dy + dz*dz + dw*dw;
    __shared__ float red[128];
    red[t] = s; __syncthreads();
    for (int o = 64; o > 0; o >>= 1) { if (t < o) red[t] += red[t+o]; __syncthreads(); }
    if (t == 0) { rowsum[row] = red[0]; idx_out[row] = (float)code; }
}

// ---------------- kernel 4: deterministic loss reduction ----------------
__global__ void vq_loss(const float* __restrict__ rowsum, float* __restrict__ loss_out) {
    const int t = threadIdx.x;        // 256
    float s = 0.f;
    for (int i = t; i < NROWS; i += 256) s += rowsum[i];   // fixed order
    __shared__ float red[256];
    red[t] = s; __syncthreads();
    for (int o = 128; o > 0; o >>= 1) { if (t < o) red[t] += red[t+o]; __syncthreads(); }
    if (t == 0) loss_out[0] = (1.0f + BETA) * red[0] / (float)((long)NROWS * D);
}

// ---------------- host launch ----------------
extern "C" void kernel_launch(void* const* d_in, const int* in_sizes, int n_in,
                              void* d_out, int out_size, void* d_ws, size_t ws_size,
                              hipStream_t stream) {
    const float* z = (const float*)d_in[0];           // [16,4096,512]
    const float* e = (const float*)d_in[1];           // [1024,512]
    float* out = (float*)d_out;                       // [z_q | loss | indices]

    // workspace layout
    char* ws = (char*)d_ws;
    unsigned* ebf    = (unsigned*)(ws);                         // 1,048,576 B
    float*    ssq    = (float*)(ws + 1048576);                  //     4,096 B
    int*      idx    = (int*)(ws + 1052672);                    //   262,144 B
    float*    rowsum = (float*)(ws + 1314816);                  //   262,144 B

    float* zq_out   = out;
    float* loss_out = out + (long)NROWS * D;
    float* idxf_out = out + (long)NROWS * D + 1;

    vq_prep<<<KCODES, 128, 0, stream>>>(e, ebf, ssq);
    vq_argmax<<<NROWS / ROWS_PB, 256, 0, stream>>>(z, ebf, ssq, idx);
    vq_gather<<<NROWS, 128, 0, stream>>>(z, e, idx, zq_out, idxf_out, rowsum);
    vq_loss<<<1, 256, 0, stream>>>(rowsum, loss_out);
}
